// RWKVBlock_13752485281860
// MI455X (gfx1250) — compile-verified, matrix operand order checked by reading, not run on hardware
//
#include <hip/hip_runtime.h>
#include <hip/hip_bf16.h>

// ---------------------------------------------------------------------------
// RWKV-v4 block for gfx1250 (MI455X).
//  - bf16 WMMA GEMMs (v_wmma_f32_16x16x32_bf16), 128x64x32 block tiles
//  - weights pre-converted fp32->bf16 once per launch; B tiles DMA'd straight
//    into LDS with global_load_async_to_lds_b128 (ASYNCcnt path),
//    double-buffered so the DMA overlaps the WMMA stage (1 barrier / K-step)
//  - fused time-shift token mixing in the A-tile loader (VALU->LDS)
//  - fused sigmoid / relu^2 / residual epilogues
//  - sequential WKV scan parallelized over B*C channels with global prefetch
// ---------------------------------------------------------------------------

typedef __attribute__((ext_vector_type(16))) __bf16 v16bf;
typedef __attribute__((ext_vector_type(8)))  float  v8f;

#define RWKV_B 8
#define RWKV_T 2048
#define RWKV_C 2048
#define RWKV_H 8192
#define RWKV_M (RWKV_B * RWKV_T)

__device__ __forceinline__ unsigned short f32_to_bf16(float f) {
    unsigned u = __builtin_bit_cast(unsigned, f);
    u += 0x7fffu + ((u >> 16) & 1u);   // round-to-nearest-even
    return (unsigned short)(u >> 16);
}

__device__ __forceinline__ uint4 pack8_bf16(const float* v) {
    union { unsigned short s[8]; uint4 q; } u;
#pragma unroll
    for (int j = 0; j < 8; ++j) u.s[j] = f32_to_bf16(v[j]);
    return u.q;
}

// Async DMA: 16 bytes global -> LDS, tracked by ASYNCcnt.
// Low 32 bits of a flat shared-pointer are the LDS byte offset (ISA 10.2).
__device__ __forceinline__ void async_copy_b128(unsigned lds_off, const void* gaddr) {
    asm volatile("global_load_async_to_lds_b128 %0, %1, off"
                 :: "v"(lds_off), "v"((unsigned long long)(size_t)gaddr)
                 : "memory");
}
__device__ __forceinline__ void wait_async0() {
    asm volatile("s_wait_asynccnt 0x0" ::: "memory");
}

// ---------------------------------------------------------------------------
// fp32 -> bf16 conversion (8 elements per thread)
// ---------------------------------------------------------------------------
__global__ __launch_bounds__(256)
void rwkv_cvt_bf16(const float* __restrict__ src, unsigned short* __restrict__ dst,
                   long n8) {
    long i = (long)blockIdx.x * 256 + threadIdx.x;
    if (i >= n8) return;
    const float4 f0 = ((const float4*)src)[2 * i];
    const float4 f1 = ((const float4*)src)[2 * i + 1];
    float v[8] = {f0.x, f0.y, f0.z, f0.w, f1.x, f1.y, f1.z, f1.w};
    ((uint4*)dst)[i] = pack8_bf16(v);
}

// ---------------------------------------------------------------------------
// LayerNorm: one 256-thread block per row of length C.
// ---------------------------------------------------------------------------
__global__ __launch_bounds__(256)
void rwkv_layernorm(const float* __restrict__ x, const float* __restrict__ w,
                    const float* __restrict__ b, float* __restrict__ y, int C) {
    const int row = blockIdx.x;
    const int tid = threadIdx.x;
    const float* xr = x + (size_t)row * C;
    float s = 0.f, s2 = 0.f;
    for (int c = tid; c < C; c += 256) { float v = xr[c]; s += v; s2 += v * v; }
    __shared__ float red[256], red2[256];
    red[tid] = s; red2[tid] = s2; __syncthreads();
    for (int off = 128; off > 0; off >>= 1) {
        if (tid < off) { red[tid] += red[tid + off]; red2[tid] += red2[tid + off]; }
        __syncthreads();
    }
    const float mean = red[0] / C;
    const float var  = red2[0] / C - mean * mean;
    const float inv  = rsqrtf(var + 1e-5f);
    float* yr = y + (size_t)row * C;
    for (int c = tid; c < C; c += 256)
        yr[c] = (xr[c] - mean) * inv * w[c] + b[c];
}

// ---------------------------------------------------------------------------
// WMMA GEMM: D[M,N] = A[M,K] * W[N,K]^T   (einsum 'mk,nk->mn'), W is bf16.
//   AMODE 1: A fp32 with fused time-shift mix (VALU convert -> ds_store)
//   AMODE 0: A fp32 (VALU convert -> ds_store)
//   AMODE 2: A bf16 (async DMA global -> LDS)
//   EPI 0: store fp32        EPI 1: sigmoid -> fp32
//   EPI 2: relu()^2 -> bf16  EPI 3: add aux (residual) -> fp32
// Block tile 128x64x32, 8 wave32s, each wave 32x32 (2x2 WMMA accumulators).
// Double-buffered LDS: DMA/stage tile i+1 while WMMAs consume tile i.
// ---------------------------------------------------------------------------
template <int AMODE, int EPI>
__global__ __launch_bounds__(256)
void rwkv_gemm_wmma(const void* __restrict__ Aptr,
                    const unsigned short* __restrict__ Wptr,
                    void* __restrict__ Dptr, const float* __restrict__ mix,
                    const float* __restrict__ aux, int M, int N, int K, int T) {
    constexpr int BM = 128, BN = 64, BK = 32, LDT = 40; // LDS row stride (pad)
    __shared__ __align__(16) unsigned short As[2][BM * LDT];
    __shared__ __align__(16) unsigned short Bs[2][BN * LDT];

    const int tid  = threadIdx.x;
    const int lane = tid & 31;
    const int wid  = tid >> 5;
    const int bm   = blockIdx.x * BM;
    const int bn   = blockIdx.y * BN;
    const int wm   = (wid & 3) << 5;   // wave row offset within block tile
    const int wn   = (wid >> 2) << 5;  // wave col offset within block tile

    v8f acc[2][2] = {};

    // Stage one K-tile (k0) into LDS buffers Asb/Bsb.
    auto stage = [&](int k0, unsigned short* Asb, unsigned short* Bsb) {
        // B tile (64x32 bf16): async DMA straight into LDS
        {
            const int r = tid >> 2, cb = (tid & 3) * 8;
            async_copy_b128((unsigned)(size_t)&Bsb[r * LDT + cb],
                            Wptr + (size_t)(bn + r) * K + k0 + cb);
        }
        if (AMODE == 2) {
            const unsigned short* A16 = (const unsigned short*)Aptr;
#pragma unroll
            for (int i = 0; i < 2; ++i) {
                int q = tid + i * 256;           // 512 chunks of 8 elements
                int r = q >> 2, cb = (q & 3) * 8;
                async_copy_b128((unsigned)(size_t)&Asb[r * LDT + cb],
                                A16 + (size_t)(bm + r) * K + k0 + cb);
            }
        } else {
            const float* A32 = (const float*)Aptr;
#pragma unroll
            for (int i = 0; i < 2; ++i) {
                int q = tid + i * 256;
                int r = q >> 2, cb = (q & 3) * 8;
                int row = bm + r;
                const float* src = A32 + (size_t)row * K + k0 + cb;
                float4 f0 = *(const float4*)(src);
                float4 f1 = *(const float4*)(src + 4);
                float v[8] = {f0.x, f0.y, f0.z, f0.w, f1.x, f1.y, f1.z, f1.w};
                if (AMODE == 1) {
                    float p[8] = {0.f, 0.f, 0.f, 0.f, 0.f, 0.f, 0.f, 0.f};
                    if ((row % T) != 0) {
                        const float* ps = A32 + (size_t)(row - 1) * K + k0 + cb;
                        float4 p0 = *(const float4*)(ps);
                        float4 p1 = *(const float4*)(ps + 4);
                        p[0] = p0.x; p[1] = p0.y; p[2] = p0.z; p[3] = p0.w;
                        p[4] = p1.x; p[5] = p1.y; p[6] = p1.z; p[7] = p1.w;
                    }
                    const float4 m0 = *(const float4*)(mix + k0 + cb);
                    const float4 m1 = *(const float4*)(mix + k0 + cb + 4);
                    float mm[8] = {m0.x, m0.y, m0.z, m0.w, m1.x, m1.y, m1.z, m1.w};
#pragma unroll
                    for (int j = 0; j < 8; ++j)
                        v[j] = v[j] * mm[j] + p[j] * (1.f - mm[j]);
                }
                *(uint4*)(&Asb[r * LDT + cb]) = pack8_bf16(v);
            }
        }
    };

    // Prologue: stage tile 0, wait, publish.
    stage(0, As[0], Bs[0]);
    wait_async0();
    __syncthreads();

    int cur = 0;
    for (int k0 = 0; k0 < K; k0 += BK) {
        const int nxt = cur ^ 1;
        // Kick off DMA / staging for the next K-tile while we compute.
        if (k0 + BK < K) stage(k0 + BK, As[nxt], Bs[nxt]);

        // ---- compute: per-wave 2x2 tiles of v_wmma_f32_16x16x32_bf16 ----
        const int rsel = lane & 15;
        const int kb   = (lane >> 4) * 8;  // lanes 0-15: K 0..7/16..23; 16-31: 8..15/24..31
#pragma unroll
        for (int mi = 0; mi < 2; ++mi) {
            union { uint4 q[2]; v16bf v; } fa;
            const int ar = wm + mi * 16 + rsel;
            fa.q[0] = *(const uint4*)(&As[cur][ar * LDT + kb]);
            fa.q[1] = *(const uint4*)(&As[cur][ar * LDT + 16 + kb]);
#pragma unroll
            for (int ni = 0; ni < 2; ++ni) {
                union { uint4 q[2]; v16bf v; } fb;
                const int bc = wn + ni * 16 + rsel;
                fb.q[0] = *(const uint4*)(&Bs[cur][bc * LDT + kb]);
                fb.q[1] = *(const uint4*)(&Bs[cur][bc * LDT + 16 + kb]);
                acc[mi][ni] = __builtin_amdgcn_wmma_f32_16x16x32_bf16(
                    false, fa.v, false, fb.v, (short)0, acc[mi][ni], false, false);
            }
        }

        // One barrier per K-step: publishes writes to nxt AND protects cur
        // (next iteration's stage writes come after this barrier).
        wait_async0();
        __syncthreads();
        cur = nxt;
    }

    // ---- epilogue: D VGPR r holds M = r + 8*(lane>=16), N = lane&15 ----
#pragma unroll
    for (int mi = 0; mi < 2; ++mi) {
#pragma unroll
        for (int ni = 0; ni < 2; ++ni) {
            const int cn    = bn + wn + ni * 16 + (lane & 15);
            const int rbase = bm + wm + mi * 16 + ((lane >> 4) << 3);
#pragma unroll
            for (int r = 0; r < 8; ++r) {
                const float x = acc[mi][ni][r];
                const size_t idx = (size_t)(rbase + r) * N + cn;
                if (EPI == 0) {
                    ((float*)Dptr)[idx] = x;
                } else if (EPI == 1) {
                    ((float*)Dptr)[idx] = 1.f / (1.f + __expf(-x));
                } else if (EPI == 2) {
                    float t = fmaxf(x, 0.f);
                    ((unsigned short*)Dptr)[idx] = f32_to_bf16(t * t);
                } else {
                    ((float*)Dptr)[idx] = x + aux[idx];
                }
            }
        }
    }
}

// ---------------------------------------------------------------------------
// WKV recurrence: one thread per (b,c) channel, sequential over T.
// y overwrites k in place. Prefetch 16 timesteps ahead (global_prefetch_b8).
// ---------------------------------------------------------------------------
__global__ __launch_bounds__(256)
void rwkv_wkv(const float* __restrict__ wdec, const float* __restrict__ ufirst,
              float* __restrict__ kbuf, const float* __restrict__ vbuf,
              int B, int T, int C) {
    const int g = blockIdx.x * 256 + threadIdx.x;   // 0 .. B*C-1
    const int b = g / C, c = g % C;
    const float w = -__expf(wdec[c]);
    const float u = ufirst[c];
    float aa = 0.f, bb = 0.f, pp = -1e38f;
    size_t idx = (size_t)b * T * C + c;
    for (int t = 0; t < T; ++t, idx += C) {
        if ((t & 15) == 0 && t + 16 < T) {
            __builtin_prefetch(kbuf + idx + (size_t)16 * C, 0, 0);
            __builtin_prefetch(vbuf + idx + (size_t)16 * C, 0, 0);
        }
        const float kt = kbuf[idx], vt = vbuf[idx];
        const float ww = u + kt;
        const float p  = fmaxf(pp, ww);
        const float e1 = __expf(pp - p);
        const float e2 = __expf(ww - p);
        kbuf[idx] = (e1 * aa + e2 * vt) / (e1 * bb + e2);
        const float ww2 = pp + w;
        const float p2  = fmaxf(ww2, kt);
        const float e1b = __expf(ww2 - p2);
        const float e2b = __expf(kt - p2);
        aa = e1b * aa + e2b * vt;
        bb = e1b * bb + e2b;
        pp = p2;
    }
}

// ---------------------------------------------------------------------------
// Elementwise helpers (float4-vectorized)
// ---------------------------------------------------------------------------
__global__ __launch_bounds__(256)
void rwkv_ew_mul(float* __restrict__ a, const float* __restrict__ b, int n4) {
    int i = blockIdx.x * 256 + threadIdx.x;
    if (i >= n4) return;
    float4 va = ((const float4*)a)[i];
    float4 vb = ((const float4*)b)[i];
    va.x *= vb.x; va.y *= vb.y; va.z *= vb.z; va.w *= vb.w;
    ((float4*)a)[i] = va;
}

__global__ __launch_bounds__(256)
void rwkv_ew_final(const float* __restrict__ x1, const float* __restrict__ r,
                   const float* __restrict__ v, float* __restrict__ out, int n4) {
    int i = blockIdx.x * 256 + threadIdx.x;
    if (i >= n4) return;
    float4 a = ((const float4*)x1)[i];
    float4 b = ((const float4*)r)[i];
    float4 c = ((const float4*)v)[i];
    a.x += b.x * c.x; a.y += b.y * c.y; a.z += b.z * c.z; a.w += b.w * c.w;
    ((float4*)out)[i] = a;
}

// ---------------------------------------------------------------------------
// Orchestration
// ---------------------------------------------------------------------------
extern "C" void kernel_launch(void* const* d_in, const int* in_sizes, int n_in,
                              void* d_out, int out_size, void* d_ws, size_t ws_size,
                              hipStream_t stream) {
    (void)in_sizes; (void)n_in; (void)out_size; (void)ws_size;
    const int B = RWKV_B, T = RWKV_T, C = RWKV_C, H = RWKV_H, M = RWKV_M;

    const float* x    = (const float*)d_in[0];
    const float* ln1w = (const float*)d_in[1];
    const float* ln1b = (const float*)d_in[2];
    const float* ln2w = (const float*)d_in[3];
    const float* ln2b = (const float*)d_in[4];
    const float* tdec = (const float*)d_in[5];
    const float* tfir = (const float*)d_in[6];
    const float* tmk  = (const float*)d_in[7];
    const float* tmv  = (const float*)d_in[8];
    const float* tmr  = (const float*)d_in[9];
    const float* Wk   = (const float*)d_in[10];
    const float* Wv   = (const float*)d_in[11];
    const float* Wr   = (const float*)d_in[12];
    const float* Wo   = (const float*)d_in[13];
    const float* cmk  = (const float*)d_in[14];
    const float* cmr  = (const float*)d_in[15];
    const float* Wck  = (const float*)d_in[16];
    const float* Wcv  = (const float*)d_in[17];
    const float* Wcr  = (const float*)d_in[18];
    float* out = (float*)d_out;

    // workspace layout (~1.0 GB)
    const size_t SZ  = (size_t)M * C * sizeof(float);            // 128 MiB
    const size_t CC  = (size_t)C * C;                            // weight elems
    const size_t CH  = (size_t)C * H;
    char* ws = (char*)d_ws;
    float* XN = (float*)(ws + 0 * SZ);   // ln1 out, later ln2 out
    float* KB = (float*)(ws + 1 * SZ);   // k -> wkv y -> sr*att
    float* VB = (float*)(ws + 2 * SZ);   // v -> kv
    float* RB = (float*)(ws + 3 * SZ);   // sigmoid(r) -> sigmoid(cr)
    float* X1 = (float*)(ws + 4 * SZ);   // x + attention out
    unsigned short* HB  = (unsigned short*)(ws + 5 * SZ);        // bf16 [M,H]
    unsigned short* WBF = (unsigned short*)(ws + 5 * SZ + (size_t)M * H * 2);
    unsigned short* wkb  = WBF + 0 * CC;
    unsigned short* wvb  = WBF + 1 * CC;
    unsigned short* wrb  = WBF + 2 * CC;
    unsigned short* wob  = WBF + 3 * CC;
    unsigned short* wcrb = WBF + 4 * CC;
    unsigned short* wckb = WBF + 5 * CC;
    unsigned short* wcvb = WBF + 5 * CC + CH;

    const dim3 blk(256);
    const dim3 gLN(M);
    const dim3 gC(M / 128, C / 64);     // GEMMs with N=C
    const dim3 gH(M / 128, H / 64);     // GEMM with N=H
    const dim3 gCvtC((unsigned)(CC / 8 / 256));
    const dim3 gCvtH((unsigned)(CH / 8 / 256));
    const int  n4 = (M * C) / 4;
    const dim3 gEW((n4 + 255) / 256);

    // 0. weights -> bf16 (halves B-operand bytes; enables async LDS DMA)
    rwkv_cvt_bf16<<<gCvtC, blk, 0, stream>>>(Wk,  wkb,  (long)(CC / 8));
    rwkv_cvt_bf16<<<gCvtC, blk, 0, stream>>>(Wv,  wvb,  (long)(CC / 8));
    rwkv_cvt_bf16<<<gCvtC, blk, 0, stream>>>(Wr,  wrb,  (long)(CC / 8));
    rwkv_cvt_bf16<<<gCvtC, blk, 0, stream>>>(Wo,  wob,  (long)(CC / 8));
    rwkv_cvt_bf16<<<gCvtC, blk, 0, stream>>>(Wcr, wcrb, (long)(CC / 8));
    rwkv_cvt_bf16<<<gCvtH, blk, 0, stream>>>(Wck, wckb, (long)(CH / 8));
    rwkv_cvt_bf16<<<gCvtH, blk, 0, stream>>>(Wcv, wcvb, (long)(CH / 8));

    // 1. xn = LN1(x)
    rwkv_layernorm<<<gLN, blk, 0, stream>>>(x, ln1w, ln1b, XN, C);
    // 2. k, v, sigmoid(r) with fused time-shift mix
    rwkv_gemm_wmma<1, 0><<<gC, blk, 0, stream>>>(XN, wkb, KB, tmk, nullptr, M, C, C, T);
    rwkv_gemm_wmma<1, 0><<<gC, blk, 0, stream>>>(XN, wvb, VB, tmv, nullptr, M, C, C, T);
    rwkv_gemm_wmma<1, 1><<<gC, blk, 0, stream>>>(XN, wrb, RB, tmr, nullptr, M, C, C, T);
    // 3. WKV scan: y overwrites KB
    rwkv_wkv<<<dim3((B * C) / 256), blk, 0, stream>>>(tdec, tfir, KB, VB, B, T, C);
    // 4. a2 = sr * y  (in place in KB)
    rwkv_ew_mul<<<gEW, blk, 0, stream>>>(KB, RB, n4);
    // 5. x1 = x + a2 * Wo^T
    rwkv_gemm_wmma<0, 3><<<gC, blk, 0, stream>>>(KB, wob, X1, nullptr, x, M, C, C, T);
    // 6. xn2 = LN2(x1) (reuse XN)
    rwkv_layernorm<<<gLN, blk, 0, stream>>>(X1, ln2w, ln2b, XN, C);
    // 7. h = relu(ck * Wck^T)^2 -> bf16, fused cmk mix
    rwkv_gemm_wmma<1, 2><<<gH, blk, 0, stream>>>(XN, wckb, HB, cmk, nullptr, M, H, C, T);
    // 8. kv = h * Wcv^T  (bf16 A via async DMA, K=H) -> VB
    rwkv_gemm_wmma<2, 0><<<gC, blk, 0, stream>>>(HB, wcvb, VB, nullptr, nullptr, M, C, H, T);
    // 9. sigmoid(cr * Wcr^T), fused cmr mix -> RB
    rwkv_gemm_wmma<1, 1><<<gC, blk, 0, stream>>>(XN, wcrb, RB, cmr, nullptr, M, C, C, T);
    // 10. out = x1 + RB * VB
    rwkv_ew_final<<<gEW, blk, 0, stream>>>(X1, RB, VB, out, n4);
}
